// Mobile2Former_83485574300254
// MI455X (gfx1250) — matrix operand order repaired and checked by reading, not compiled, not run on hardware
//
#include <hip/hip_runtime.h>
#include <hip/hip_bf16.h>

typedef __attribute__((ext_vector_type(16))) __bf16       v16bf;
typedef __attribute__((ext_vector_type(8)))  float        v8f;
typedef __attribute__((ext_vector_type(8)))  unsigned int v8u;

#define B_      128
#define C_      64
#define HW_     3136
#define M_      6
#define D_      192
#define HEADS_  8
#define INNER_  512
#define ROWS_   48              // HEADS_ * M_
#define NTILE   32
#define NT_CNT  (HW_ / NTILE)   // 98
#define WAVES   8
#define NPART   4               // N-split workgroups per batch
#define PART_STRIDE (ROWS_ + ROWS_ + ROWS_ * C_)   // m[48] + l[48] + O[48*64] floats
#define NEG_BIG (-3.0e38f)

__device__ __forceinline__ __bf16 f2bf(float f) {
    return (__bf16)f;           // native v_cvt_pk_bf16_f32 on gfx1250
}

// 16x32 bf16 A-fragment from a 32-element K-slice in LDS: per the CDNA5 layout,
// lane (l16,half) holds K = {half*8+0..7, 16+half*8+0..7} of row M=l16
// -> two contiguous 16B chunks => two ds_load_b128.
__device__ __forceinline__ v16bf ld_afrag32(const __bf16* slice, int half) {
    const uint4* q = (const uint4*)slice;
    uint4 c0 = q[half];
    uint4 c1 = q[2 + half];
    v8u t;
    t[0] = c0.x; t[1] = c0.y; t[2] = c0.z; t[3] = c0.w;
    t[4] = c1.x; t[5] = c1.y; t[6] = c1.z; t[7] = c1.w;
    return __builtin_bit_cast(v16bf, t);
}

// Phases 1-3 of the flash attention for one (batch, N-part).
// On exit (after an internal barrier): lds_res = unnormalized O partial
// (rescaled to the block-level row max), lds_M / lds_L = block row stats.
template <int PARTS>
__device__ __forceinline__ void flash_core(
    const float* __restrict__ X, const float* __restrict__ Zb,
    const float* __restrict__ Wq, const float* __restrict__ bq, int part,
    __bf16* lds_q, __bf16* lds_p, float* lds_res,
    float* lds_mw, float* lds_lw, float* lds_M, float* lds_L)
{
    const int tid  = threadIdx.x;
    const int wave = tid >> 5;
    const int lane = tid & 31;
    const int l16  = lane & 15;
    const int half = lane >> 4;

    // ---------- Phase 1: Q = (z @ Wq + bq) * scale -> LDS bf16 ; zero lds_res
    for (int e = tid; e < ROWS_ * C_; e += 256) {
        int m = e >> 9;
        int i = e & (INNER_ - 1);
        float acc = bq[i];
        const float* zr = Zb + m * D_;
        const float* wc = Wq + i;
        #pragma unroll 4
        for (int d = 0; d < D_; ++d)
            acc += zr[d] * wc[(size_t)d * INNER_];
        lds_q[e]   = f2bf(acc * 0.125f);     // fold in C^-0.5
        lds_res[e] = 0.0f;
    }
    __syncthreads();

    // ---------- Flash state
    v8f acc_o[3][4];
    #pragma unroll
    for (int mt = 0; mt < 3; ++mt)
        #pragma unroll
        for (int ct = 0; ct < 4; ++ct)
            #pragma unroll
            for (int r = 0; r < 8; ++r) acc_o[mt][ct][r] = 0.0f;

    float m_s[3][8], l_s[3][8];
    #pragma unroll
    for (int mt = 0; mt < 3; ++mt)
        #pragma unroll
        for (int r = 0; r < 8; ++r) { m_s[mt][r] = NEG_BIG; l_s[mt][r] = 0.0f; }

    __bf16* pbuf = lds_p + wave * (3 * 16 * NTILE);

    // ---------- Phase 2: this part's N tiles, round-robin across waves
    for (int t = part + PARTS * wave; t < NT_CNT; t += PARTS * WAVES) {
        const int nbase = t * NTILE;

        if (t + PARTS * WAVES < NT_CNT) {
            const float* pf = X + (size_t)lane * HW_
                                + (size_t)(t + PARTS * WAVES) * NTILE;
            __builtin_prefetch(pf, 0, 0);
            __builtin_prefetch(pf + (size_t)32 * HW_, 0, 0);
        }

        // K-side B fragments (K = channel c, N = column n)
        v16bf bb[2][2];
        #pragma unroll
        for (int ks = 0; ks < 2; ++ks)
            #pragma unroll
            for (int nc = 0; nc < 2; ++nc) {
                int n = nbase + nc * 16 + l16;
                #pragma unroll
                for (int e = 0; e < 16; ++e) {
                    int c = ks * 32 + half * 8 + e + (e & 8);
                    bb[ks][nc][e] = f2bf(X[(size_t)c * HW_ + n]);
                }
            }

        // S = Q @ K
        v8f sacc[3][2];
        #pragma unroll
        for (int mt = 0; mt < 3; ++mt) {
            const __bf16* qrow = &lds_q[(mt * 16 + l16) * C_];
            v16bf a0 = ld_afrag32(qrow,      half);
            v16bf a1 = ld_afrag32(qrow + 32, half);
            #pragma unroll
            for (int nc = 0; nc < 2; ++nc) {
                v8f cz;
                #pragma unroll
                for (int r = 0; r < 8; ++r) cz[r] = 0.0f;
                cz = __builtin_amdgcn_wmma_f32_16x16x32_bf16(
                         false, a0, false, bb[0][nc], (short)0, cz, false, false);
                sacc[mt][nc] = __builtin_amdgcn_wmma_f32_16x16x32_bf16(
                         false, a1, false, bb[1][nc], (short)0, cz, false, false);
            }
        }

        // V-side B fragments (K = column n, N = channel c)
        v16bf vb[4];
        #pragma unroll
        for (int ct = 0; ct < 4; ++ct) {
            int c = ct * 16 + l16;
            const float* xr = X + (size_t)c * HW_ + nbase + half * 8;
            #pragma unroll
            for (int e = 0; e < 16; e += 2) {
                float2 two = *(const float2*)(xr + e + (e & 8));
                vb[ct][e]     = f2bf(two.x);
                vb[ct][e + 1] = f2bf(two.y);
            }
        }

        // online softmax (rows live in 16-lane halves)
        #pragma unroll
        for (int mt = 0; mt < 3; ++mt) {
            #pragma unroll
            for (int r = 0; r < 8; ++r) {
                float s0 = sacc[mt][0][r];
                float s1 = sacc[mt][1][r];
                float mx = fmaxf(s0, s1);
                #pragma unroll
                for (int off = 8; off >= 1; off >>= 1)
                    mx = fmaxf(mx, __shfl_xor(mx, off, 32));
                float mold = m_s[mt][r];
                float mnew = fmaxf(mold, mx);
                float corr = __expf(mold - mnew);
                float p0 = __expf(s0 - mnew);
                float p1 = __expf(s1 - mnew);
                float ps = p0 + p1;
                #pragma unroll
                for (int off = 8; off >= 1; off >>= 1)
                    ps += __shfl_xor(ps, off, 32);
                l_s[mt][r] = l_s[mt][r] * corr + ps;
                m_s[mt][r] = mnew;
                #pragma unroll
                for (int ct = 0; ct < 4; ++ct)
                    acc_o[mt][ct][r] *= corr;
                int lrow = r + half * 8;
                __bf16* pp = pbuf + (mt * 16 + lrow) * NTILE;
                pp[l16]      = f2bf(p0);
                pp[16 + l16] = f2bf(p1);
            }
        }

        // O += P @ V
        #pragma unroll
        for (int mt = 0; mt < 3; ++mt) {
            v16bf pa = ld_afrag32(pbuf + (mt * 16 + l16) * NTILE, half);
            #pragma unroll
            for (int ct = 0; ct < 4; ++ct)
                acc_o[mt][ct] = __builtin_amdgcn_wmma_f32_16x16x32_bf16(
                        false, pa, false, vb[ct], (short)0, acc_o[mt][ct], false, false);
        }
    }

    // ---------- Phase 3: merge the 8 waves of this block
    if (l16 == 0) {
        #pragma unroll
        for (int mt = 0; mt < 3; ++mt)
            #pragma unroll
            for (int r = 0; r < 8; ++r) {
                int row = mt * 16 + r + half * 8;
                lds_mw[wave * ROWS_ + row] = m_s[mt][r];
                lds_lw[wave * ROWS_ + row] = l_s[mt][r];
            }
    }
    __syncthreads();

    if (tid < ROWS_) {
        float Mg = NEG_BIG;
        for (int w = 0; w < WAVES; ++w) Mg = fmaxf(Mg, lds_mw[w * ROWS_ + tid]);
        float Lg = 0.0f;
        for (int w = 0; w < WAVES; ++w)
            Lg += __expf(lds_mw[w * ROWS_ + tid] - Mg) * lds_lw[w * ROWS_ + tid];
        lds_M[tid] = Mg;
        lds_L[tid] = Lg;
    }
    __syncthreads();

    #pragma unroll
    for (int mt = 0; mt < 3; ++mt)
        #pragma unroll
        for (int r = 0; r < 8; ++r) {
            int row = mt * 16 + r + half * 8;
            float f = __expf(m_s[mt][r] - lds_M[row]);
            #pragma unroll
            for (int ct = 0; ct < 4; ++ct) {
                int c = ct * 16 + l16;
                atomicAdd(&lds_res[row * C_ + c], acc_o[mt][ct][r] * f);
            }
        }
    __syncthreads();
}

// Shared output projection: lds_res holds normalized res48 [48][64].
__device__ __forceinline__ void out_proj(
    const float* __restrict__ Zb, const float* __restrict__ Wo,
    const float* __restrict__ bo, const float* lds_res, float* outb)
{
    for (int e = threadIdx.x; e < M_ * D_; e += 256) {
        int m = e / D_;
        int j = e - m * D_;
        float acc = bo[j];
        for (int i = 0; i < INNER_; ++i) {
            int h = i >> 6;
            int c = i & 63;
            acc += lds_res[(h * M_ + m) * C_ + c] * Wo[(size_t)i * D_ + j];
        }
        outb[e] = acc + Zb[e];
    }
}

#define DECL_SHARED() \
    __shared__ __bf16 lds_q[ROWS_ * C_]; \
    __shared__ __bf16 lds_p[WAVES * 3 * 16 * NTILE]; \
    __shared__ float  lds_res[ROWS_ * C_]; \
    __shared__ float  lds_mw[WAVES * ROWS_]; \
    __shared__ float  lds_lw[WAVES * ROWS_]; \
    __shared__ float  lds_M[ROWS_]; \
    __shared__ float  lds_L[ROWS_]

// ---------- Path A: NPART workgroups per batch -> partials in d_ws
__global__ __launch_bounds__(256, 1)
void m2f_part_kernel(const float* __restrict__ x,  const float* __restrict__ z,
                     const float* __restrict__ Wq, const float* __restrict__ bq,
                     float* __restrict__ ws)
{
    DECL_SHARED();
    const int b    = blockIdx.x / NPART;
    const int part = blockIdx.x % NPART;
    const float* X  = x + (size_t)b * C_ * HW_;
    const float* Zb = z + (size_t)b * M_ * D_;

    flash_core<NPART>(X, Zb, Wq, bq, part,
                      lds_q, lds_p, lds_res, lds_mw, lds_lw, lds_M, lds_L);

    float* wsp = ws + (size_t)(b * NPART + part) * PART_STRIDE;
    for (int e = threadIdx.x; e < ROWS_; e += 256) {
        wsp[e]         = lds_M[e];
        wsp[ROWS_ + e] = lds_L[e];
    }
    for (int e = threadIdx.x; e < ROWS_ * C_; e += 256)
        wsp[2 * ROWS_ + e] = lds_res[e];
}

__global__ __launch_bounds__(256, 1)
void m2f_merge_kernel(const float* __restrict__ z,  const float* __restrict__ Wo,
                      const float* __restrict__ bo, const float* __restrict__ ws,
                      float* __restrict__ out)
{
    __shared__ float lds_res[ROWS_ * C_];
    __shared__ float lds_w[NPART * ROWS_];
    __shared__ float lds_L[ROWS_];
    const int b   = blockIdx.x;
    const int tid = threadIdx.x;
    const float* base = ws + (size_t)b * NPART * PART_STRIDE;

    if (tid < ROWS_) {
        float Mg = NEG_BIG;
        #pragma unroll
        for (int p = 0; p < NPART; ++p)
            Mg = fmaxf(Mg, base[p * PART_STRIDE + tid]);
        float Lg = 0.0f;
        #pragma unroll
        for (int p = 0; p < NPART; ++p) {
            float w = __expf(base[p * PART_STRIDE + tid] - Mg);
            lds_w[p * ROWS_ + tid] = w;
            Lg += w * base[p * PART_STRIDE + ROWS_ + tid];
        }
        lds_L[tid] = (Lg > 0.0f) ? Lg : 1.0f;
    }
    __syncthreads();

    for (int e = tid; e < ROWS_ * C_; e += 256) {
        int row = e >> 6;
        float acc = 0.0f;
        #pragma unroll
        for (int p = 0; p < NPART; ++p)
            acc += lds_w[p * ROWS_ + row] * base[p * PART_STRIDE + 2 * ROWS_ + e];
        lds_res[e] = acc / lds_L[row];
    }
    __syncthreads();

    out_proj(z + (size_t)b * M_ * D_, Wo, bo, lds_res,
             out + (size_t)b * M_ * D_);
}

// ---------- Path B fallback: one workgroup per batch (no workspace needed)
__global__ __launch_bounds__(256, 1)
void m2f_attn_kernel(const float* __restrict__ x,  const float* __restrict__ z,
                     const float* __restrict__ Wq, const float* __restrict__ bq,
                     const float* __restrict__ Wo, const float* __restrict__ bo,
                     float* __restrict__ out)
{
    DECL_SHARED();
    const int b = blockIdx.x;
    const float* X  = x + (size_t)b * C_ * HW_;
    const float* Zb = z + (size_t)b * M_ * D_;

    flash_core<1>(X, Zb, Wq, bq, 0,
                  lds_q, lds_p, lds_res, lds_mw, lds_lw, lds_M, lds_L);

    for (int e = threadIdx.x; e < ROWS_ * C_; e += 256) {
        float L = lds_L[e >> 6];
        lds_res[e] = lds_res[e] / ((L > 0.0f) ? L : 1.0f);
    }
    __syncthreads();

    out_proj(Zb, Wo, bo, lds_res, out + (size_t)b * M_ * D_);
}

extern "C" void kernel_launch(void* const* d_in, const int* in_sizes, int n_in,
                              void* d_out, int out_size, void* d_ws, size_t ws_size,
                              hipStream_t stream) {
    const float* x  = (const float*)d_in[0];
    const float* z  = (const float*)d_in[1];
    const float* Wq = (const float*)d_in[2];
    const float* bq = (const float*)d_in[3];
    const float* Wo = (const float*)d_in[4];
    const float* bo = (const float*)d_in[5];
    float* out = (float*)d_out;
    (void)in_sizes; (void)n_in; (void)out_size;

    const size_t need = (size_t)B_ * NPART * PART_STRIDE * sizeof(float);
    if (ws_size >= need) {
        float* ws = (float*)d_ws;
        m2f_part_kernel<<<dim3(B_ * NPART), dim3(256), 0, stream>>>(x, z, Wq, bq, ws);
        m2f_merge_kernel<<<dim3(B_), dim3(256), 0, stream>>>(z, Wo, bo, ws, out);
    } else {
        m2f_attn_kernel<<<dim3(B_), dim3(256), 0, stream>>>(x, z, Wq, bq, Wo, bo, out);
    }
}
